// MhAttention_41317585388086
// MI455X (gfx1250) — compile-verified, matrix operand order checked by reading, not run on hardware
//
#include <hip/hip_runtime.h>
#include <hip/hip_bf16.h>
#include <stdint.h>

// ---------------------------------------------------------------------------
// MI455X (gfx1250) multi-head causal attention, bf16 WMMA pipeline with
// async global->LDS staging (ASYNCcnt) and double-buffered GEMM tiles.
// ---------------------------------------------------------------------------

typedef __bf16 v16bf __attribute__((ext_vector_type(16)));
typedef float  v8f   __attribute__((ext_vector_type(8)));

#define D_MODEL  1024
#define N_HEADS  16
#define BATCH    2
#define SEQ      2048
#define HEAD_DIM 64

// ---------------------------------------------------------------------------
// Async DMA: copy 16 contiguous bytes global -> LDS per lane.
// ISA 10.x / 15.18.3: GLOBAL_LOAD_ASYNC_TO_LDS_B128 (ASYNCcnt).
// vdst = per-lane LDS byte address, vaddr = per-lane 64-bit global address.
// ---------------------------------------------------------------------------
__device__ inline void async_load_b128(void* lds_ptr, const void* gptr) {
  unsigned lds_off = (unsigned)(size_t)lds_ptr;  // flat addr low 32 bits == LDS offset
  asm volatile("global_load_async_to_lds_b128 %0, %1, off"
               :: "v"(lds_off), "v"(gptr)
               : "memory");
}
__device__ inline void wait_async() {
  asm volatile("s_wait_asynccnt 0x0" ::: "memory");
}

// ---------------------------------------------------------------------------
// Fragment loader. ISA 7.12.2, 16-bit A-matrix 16x32 layout (wave32):
//   lanes 0-15  : row = lane,    elements 0..7 -> K=k0+0..7,  8..15 -> K=k0+16..23
//   lanes 16-31 : row = lane-16, elements 0..7 -> K=k0+8..15, 8..15 -> K=k0+24..31
// Same striping (row -> column) serves B when the LDS tile is stored [n][k].
// ---------------------------------------------------------------------------
__device__ inline v16bf load_frag(const __bf16* base, int rc0, int k0, int ld) {
  const int lane = threadIdx.x & 31;
  const int half = lane >> 4;
  const int m    = rc0 + (lane & 15);
  v16bf f;
#pragma unroll
  for (int i = 0; i < 16; ++i) {
    const int k = k0 + half * 8 + ((i < 8) ? i : (i + 8));
    f[i] = base[m * ld + k];
  }
  return f;
}

__device__ inline v8f wmma_bf16(v16bf a, v16bf b, v8f c) {
  // (neg_a, A, neg_b, B, c_mod, C, reuse_a, reuse_b)
  return __builtin_amdgcn_wmma_f32_16x16x32_bf16(false, a, false, b, (short)0, c,
                                                 false, false);
}

// ---------------------------------------------------------------------------
// fp32 -> bf16 conversion
// ---------------------------------------------------------------------------
__global__ void convert_f32_bf16_kernel(const float* __restrict__ in,
                                        __bf16* __restrict__ out, int n) {
  int i = blockIdx.x * blockDim.x + threadIdx.x;
  if (i < n) out[i] = (__bf16)in[i];
}

// ---------------------------------------------------------------------------
// Tiled bf16 WMMA GEMM: C[M,N] = A[M,K] * B[K,N], row-major.
// Block tile 128x128, 8 waves; each wave owns a 32x64 sub-tile (2x4 WMMA tiles).
// Double-buffered LDS; A tile staged via async DMA, B tile transposed in VGPRs.
// ---------------------------------------------------------------------------
template <typename OutT>
__global__ __launch_bounds__(256) void gemm_bf16_kernel(
    const __bf16* __restrict__ A, const __bf16* __restrict__ B,
    OutT* __restrict__ C, int M, int N, int K) {
  __shared__ __align__(16) __bf16 As[2][128 * 32];   // [row][k]
  __shared__ __align__(16) __bf16 Bs[2][128 * 32];   // [n][k]

  const int t       = threadIdx.x;
  const int rowBase = blockIdx.y * 128;
  const int colBase = blockIdx.x * 128;
  const int wave    = t >> 5;
  const int wr      = (wave & 3) * 32;   // wave row offset in block tile
  const int wc      = (wave >> 2) * 64;  // wave col offset in block tile

  v8f zero = {};
  v8f acc[2][4];
#pragma unroll
  for (int i = 0; i < 2; ++i)
#pragma unroll
    for (int j = 0; j < 4; ++j) acc[i][j] = zero;

  // --- tile staging helpers -------------------------------------------------
  auto stageA = [&](int buf, int kk) {
#pragma unroll
    for (int c = 0; c < 2; ++c) {
      int e = (c * 256 + t) * 8;            // 8 bf16 = 16 bytes per chunk
      int r = e >> 5, k = e & 31;
      async_load_b128(&As[buf][e], &A[(size_t)(rowBase + r) * K + kk + k]);
    }
  };
  auto stageB = [&](int buf, int kk) {
#pragma unroll
    for (int j = 0; j < 16; ++j) {
      int idx = j * 256 + t;
      int k = idx >> 7, n = idx & 127;
      Bs[buf][n * 32 + k] = B[(size_t)(kk + k) * N + colBase + n];
    }
  };

  // prologue: stage first k-slab
  stageA(0, 0);
  stageB(0, 0);
  wait_async();
  __syncthreads();

  for (int kk = 0; kk < K; kk += 32) {
    const int cur = (kk >> 5) & 1;
    const int nxt = cur ^ 1;
    // kick off next slab's DMA before computing on the current one
    if (kk + 32 < K) {
      stageA(nxt, kk + 32);
      stageB(nxt, kk + 32);
      __builtin_prefetch(&B[(size_t)(kk + 64 + (t >> 3)) * N + colBase], 0, 1);
    }

    v16bf a0 = load_frag(&As[cur][0], wr, 0, 32);
    v16bf a1 = load_frag(&As[cur][0], wr + 16, 0, 32);
#pragma unroll
    for (int j = 0; j < 4; ++j) {
      v16bf b = load_frag(&Bs[cur][0], wc + j * 16, 0, 32);
      acc[0][j] = wmma_bf16(a0, b, acc[0][j]);
      acc[1][j] = wmma_bf16(a1, b, acc[1][j]);
    }

    wait_async();      // next-slab DMA complete (this wave's share)
    __syncthreads();   // all waves done reading cur / writing nxt
  }

  // C/D layout: VGPR r, lanes 0-15 -> row r, lanes 16-31 -> row r+8; col = lane&15
  const int lane  = t & 31;
  const int ccol  = lane & 15;
  const int chalf = lane >> 4;
#pragma unroll
  for (int i = 0; i < 2; ++i)
#pragma unroll
    for (int j = 0; j < 4; ++j)
#pragma unroll
      for (int r = 0; r < 8; ++r) {
        int row = rowBase + wr + i * 16 + r + chalf * 8;
        int col = colBase + wc + j * 16 + ccol;
        C[(size_t)row * N + col] = (OutT)acc[i][j][r];
      }
}

// ---------------------------------------------------------------------------
// RoPE + head split/transpose: [B*S, H*64] bf16 -> [B,H,S,64] bf16.
// One thread per (even,odd) rotation pair.
// ---------------------------------------------------------------------------
__global__ void rope_transpose_kernel(const __bf16* __restrict__ Tq,
                                      const __bf16* __restrict__ Tk,
                                      const __bf16* __restrict__ Tv,
                                      __bf16* __restrict__ Qh,
                                      __bf16* __restrict__ Kh,
                                      __bf16* __restrict__ Vh) {
  int idx = blockIdx.x * blockDim.x + threadIdx.x;  // B*S*H*32 pairs
  int j = idx & 31;
  int h = (idx >> 5) & (N_HEADS - 1);
  int s = (idx >> 9) & (SEQ - 1);
  int b = idx >> 20;

  size_t inBase  = ((size_t)b * SEQ + s) * D_MODEL + h * HEAD_DIM + 2 * j;
  size_t outBase = (((size_t)b * N_HEADS + h) * SEQ + s) * HEAD_DIM + 2 * j;

  // inv_freq = 10000^(-2j/64); ln(10000) = 9.210340371976184
  float freq = (float)s * __expf(-((float)(2 * j) * (1.0f / HEAD_DIM)) *
                                 9.210340371976184f);
  float sn, cs;
  __sincosf(freq, &sn, &cs);

  float q1 = (float)Tq[inBase], q2 = (float)Tq[inBase + 1];
  Qh[outBase]     = (__bf16)(q1 * cs - q2 * sn);
  Qh[outBase + 1] = (__bf16)(q1 * sn + q2 * cs);

  float k1 = (float)Tk[inBase], k2 = (float)Tk[inBase + 1];
  Kh[outBase]     = (__bf16)(k1 * cs - k2 * sn);
  Kh[outBase + 1] = (__bf16)(k1 * sn + k2 * cs);

  Vh[outBase]     = Tv[inBase];
  Vh[outBase + 1] = Tv[inBase + 1];
}

// ---------------------------------------------------------------------------
// Flash attention (causal, online softmax). One block = one (b,h) x 64 q-rows.
// 4 waves; each wave owns 16 q-rows. 64-key blocks; Q@K^T and P@V via WMMA.
// Q and K tiles staged via async DMA; V transposed through VGPRs.
// ---------------------------------------------------------------------------
__global__ __launch_bounds__(128) void attention_kernel(
    const __bf16* __restrict__ Qh, const __bf16* __restrict__ Kh,
    const __bf16* __restrict__ Vh, __bf16* __restrict__ merged) {
  __shared__ __align__(16) __bf16 Qs[64 * 64];    // [qrow][d]
  __shared__ __align__(16) __bf16 Ks[64 * 64];    // [key][d]  == B-frag [n=key][k=d]
  __shared__ __align__(16) __bf16 Vs[64 * 64];    // [d][key]  == B-frag [n=d][k=key]
  __shared__ __align__(16) __bf16 Ps[4][16 * 64]; // per-wave probabilities [m][key]

  const int t     = threadIdx.x;
  const int wave  = t >> 5;
  const int lane  = t & 31;
  const int half  = lane >> 4;
  const int lcol  = lane & 15;
  const int qBase = blockIdx.x * 64;
  const int h     = blockIdx.y;
  const int b     = blockIdx.z;
  const size_t bh = ((size_t)b * N_HEADS + h) * SEQ;

  // Stage Q tile once (async DMA; completion folded into first k-block wait)
#pragma unroll
  for (int c = 0; c < 4; ++c) {
    int e = (c * 128 + t) * 8;
    async_load_b128(&Qs[e], &Qh[(bh + qBase) * HEAD_DIM + e]);
  }

  v8f zero = {};
  v8f acc_o[4];
#pragma unroll
  for (int tt = 0; tt < 4; ++tt) acc_o[tt] = zero;
  float m_run[8], l_run[8];
#pragma unroll
  for (int r = 0; r < 8; ++r) { m_run[r] = -1e30f; l_run[r] = 0.0f; }

  for (int kb = 0; kb <= qBase; kb += 64) {
    // Stage K tile (async, layout already matches B-fragment [key][d])
#pragma unroll
    for (int c = 0; c < 4; ++c) {
      int e = (c * 128 + t) * 8;
      async_load_b128(&Ks[e], &Kh[(bh + kb) * HEAD_DIM + e]);
    }
    // Stage V transposed (VGPR path)
#pragma unroll
    for (int j = 0; j < 32; ++j) {
      int idx = j * 128 + t;
      int key = idx >> 6, d = idx & 63;
      Vs[d * 64 + key] = Vh[(bh + kb) * HEAD_DIM + idx];
    }
    wait_async();
    __syncthreads();

    // S = Q * K^T  (16 q-rows x 64 keys per wave)
    v16bf qf0 = load_frag(Qs, wave * 16, 0, 64);
    v16bf qf1 = load_frag(Qs, wave * 16, 32, 64);
    v8f sc[4];
#pragma unroll
    for (int tt = 0; tt < 4; ++tt) {
      v8f z = zero;
      v16bf kf0 = load_frag(Ks, tt * 16, 0, 64);
      v16bf kf1 = load_frag(Ks, tt * 16, 32, 64);
      z = wmma_bf16(qf0, kf0, z);
      z = wmma_bf16(qf1, kf1, z);
      sc[tt] = z;
    }

    // scale + causal mask (diagonal block only)
    const bool diag = (kb == qBase);
#pragma unroll
    for (int tt = 0; tt < 4; ++tt) {
      int key = kb + tt * 16 + lcol;
#pragma unroll
      for (int r = 0; r < 8; ++r) {
        float v = sc[tt][r] * 0.125f;   // 1/sqrt(64)
        if (diag) {
          int qrow = qBase + wave * 16 + r + half * 8;
          if (key > qrow) v = -1e30f;
        }
        sc[tt][r] = v;
      }
    }

    // online softmax per row (rows split across lane halves per C layout)
#pragma unroll
    for (int r = 0; r < 8; ++r) {
      float v = fmaxf(fmaxf(sc[0][r], sc[1][r]), fmaxf(sc[2][r], sc[3][r]));
#pragma unroll
      for (int off = 1; off < 16; off <<= 1) v = fmaxf(v, __shfl_xor(v, off, 16));
      float mnew  = fmaxf(m_run[r], v);
      float alpha = __expf(m_run[r] - mnew);
      m_run[r] = mnew;
      float sum = 0.0f;
#pragma unroll
      for (int tt = 0; tt < 4; ++tt) {
        float p = __expf(sc[tt][r] - mnew);
        sc[tt][r] = p;
        sum += p;
      }
#pragma unroll
      for (int off = 1; off < 16; off <<= 1) sum += __shfl_xor(sum, off, 16);
      l_run[r] = l_run[r] * alpha + sum;
#pragma unroll
      for (int tt = 0; tt < 4; ++tt) acc_o[tt][r] *= alpha;
      int lrow = r + half * 8;
#pragma unroll
      for (int tt = 0; tt < 4; ++tt)
        Ps[wave][lrow * 64 + tt * 16 + lcol] = (__bf16)sc[tt][r];
    }

    // O += P * V   (P re-fragmented via LDS; wave-local DS ops are in-order)
    v16bf pf0 = load_frag(&Ps[wave][0], 0, 0, 64);
    v16bf pf1 = load_frag(&Ps[wave][0], 0, 32, 64);
#pragma unroll
    for (int tt = 0; tt < 4; ++tt) {
      v16bf vf0 = load_frag(Vs, tt * 16, 0, 64);
      v16bf vf1 = load_frag(Vs, tt * 16, 32, 64);
      acc_o[tt] = wmma_bf16(pf0, vf0, acc_o[tt]);
      acc_o[tt] = wmma_bf16(pf1, vf1, acc_o[tt]);
    }
    __syncthreads();
  }

  // normalize and write merged [B, S, H*64]
#pragma unroll
  for (int tt = 0; tt < 4; ++tt)
#pragma unroll
    for (int r = 0; r < 8; ++r) {
      int qrow = qBase + wave * 16 + r + half * 8;
      float o = acc_o[tt][r] / l_run[r];
      merged[((size_t)b * SEQ + qrow) * D_MODEL + h * HEAD_DIM + tt * 16 + lcol] =
          (__bf16)o;
    }
}

// ---------------------------------------------------------------------------
// Host orchestration
// ---------------------------------------------------------------------------
extern "C" void kernel_launch(void* const* d_in, const int* in_sizes, int n_in,
                              void* d_out, int out_size, void* d_ws, size_t ws_size,
                              hipStream_t stream) {
  (void)in_sizes; (void)n_in; (void)out_size; (void)ws_size;

  const float* x  = (const float*)d_in[0];
  const float* qw = (const float*)d_in[1];
  const float* kw = (const float*)d_in[2];
  const float* vw = (const float*)d_in[3];
  const float* ow = (const float*)d_in[4];
  float* out = (float*)d_out;

  const size_t MT  = (size_t)BATCH * SEQ;         // 4096 token rows
  const size_t XN  = MT * D_MODEL;                // 4M elements
  const size_t WN  = (size_t)D_MODEL * D_MODEL;   // 1M elements

  char* ws = (char*)d_ws;
  size_t off = 0;
  auto alloc = [&](size_t bytes) -> char* {
    char* p = ws + off;
    off += (bytes + 255) & ~(size_t)255;
    return p;
  };
  __bf16* xb  = (__bf16*)alloc(XN * sizeof(__bf16));
  __bf16* qwb = (__bf16*)alloc(WN * sizeof(__bf16));
  __bf16* kwb = (__bf16*)alloc(WN * sizeof(__bf16));
  __bf16* vwb = (__bf16*)alloc(WN * sizeof(__bf16));
  __bf16* owb = (__bf16*)alloc(WN * sizeof(__bf16));
  __bf16* Tq  = (__bf16*)alloc(XN * sizeof(__bf16));
  __bf16* Tk  = (__bf16*)alloc(XN * sizeof(__bf16));
  __bf16* Tv  = (__bf16*)alloc(XN * sizeof(__bf16));
  __bf16* Qh  = (__bf16*)alloc(XN * sizeof(__bf16));
  __bf16* Kh  = (__bf16*)alloc(XN * sizeof(__bf16));
  __bf16* Vh  = (__bf16*)alloc(XN * sizeof(__bf16));
  __bf16* merged = Tq;  // Tq dead after RoPE; reuse for attention output

  // 1) fp32 -> bf16
  convert_f32_bf16_kernel<<<(int)(XN / 256), 256, 0, stream>>>(x, xb, (int)XN);
  convert_f32_bf16_kernel<<<(int)(WN / 256), 256, 0, stream>>>(qw, qwb, (int)WN);
  convert_f32_bf16_kernel<<<(int)(WN / 256), 256, 0, stream>>>(kw, kwb, (int)WN);
  convert_f32_bf16_kernel<<<(int)(WN / 256), 256, 0, stream>>>(vw, vwb, (int)WN);
  convert_f32_bf16_kernel<<<(int)(WN / 256), 256, 0, stream>>>(ow, owb, (int)WN);

  // 2) QKV projections (WMMA, async-staged tiles)
  dim3 gg(D_MODEL / 128, (unsigned)(MT / 128));
  gemm_bf16_kernel<__bf16><<<gg, 256, 0, stream>>>(xb, qwb, Tq, (int)MT, D_MODEL, D_MODEL);
  gemm_bf16_kernel<__bf16><<<gg, 256, 0, stream>>>(xb, kwb, Tk, (int)MT, D_MODEL, D_MODEL);
  gemm_bf16_kernel<__bf16><<<gg, 256, 0, stream>>>(xb, vwb, Tv, (int)MT, D_MODEL, D_MODEL);

  // 3) RoPE + [B,H,S,D] transpose
  int ropeThreads = BATCH * SEQ * N_HEADS * (HEAD_DIM / 2);
  rope_transpose_kernel<<<ropeThreads / 256, 256, 0, stream>>>(Tq, Tk, Tv, Qh, Kh, Vh);

  // 4) causal flash attention (WMMA)
  attention_kernel<<<dim3(SEQ / 64, N_HEADS, BATCH), 128, 0, stream>>>(Qh, Kh, Vh, merged);

  // 5) output projection (WMMA, fp32 out)
  gemm_bf16_kernel<float><<<gg, 256, 0, stream>>>(merged, owb, out, (int)MT, D_MODEL, D_MODEL);
}
